// GCN_pyg_76184129896719
// MI455X (gfx1250) — matrix operand order repaired and checked by reading, not compiled, not run on hardware
//
#include <hip/hip_runtime.h>
#include <hip/hip_bf16.h>

// ---------------------------------------------------------------------------
// GCN (3-layer, EMB=128) for gfx1250.
// GEMMs via v_wmma_f32_16x16x32_bf16 (f32 accumulate); edge aggregation via
// fused gather + global_atomic_add_f32 scatter; everything else streaming.
// ---------------------------------------------------------------------------

#define N_NODES 50000
#define N_EDGES 600000
#define N_GRAPHS 2048
#define EMB 128

typedef __attribute__((ext_vector_type(16))) __bf16 v16bf;
typedef __attribute__((ext_vector_type(8)))  float  v8f;
typedef __attribute__((ext_vector_type(8)))  unsigned int v8u;

// ------------------------- generic helpers ---------------------------------

__global__ void zero_f32_kernel(float* __restrict__ p, long long n) {
    long long i = (long long)blockIdx.x * blockDim.x + threadIdx.x;
    if (i < n) p[i] = 0.0f;
}

__global__ void cvt_f32_to_bf16_kernel(const float* __restrict__ src,
                                       __bf16* __restrict__ dst, long long n) {
    long long i = (long long)blockIdx.x * blockDim.x + threadIdx.x;
    if (i < n) dst[i] = (__bf16)src[i];
}

// W[layer,k] is [K=128][N=128] row-major. For WMMA-B fragments we want, per
// output column n, 2-consecutive-K bf16 pairs contiguous in memory -> store
// transposed: wt[lk][n][k] = bf16(W[lk][k][n]).
__global__ void prep_w_kernel(const float* __restrict__ W, __bf16* __restrict__ wt) {
    int idx = blockIdx.x * 256 + threadIdx.x;       // 9 * 128 * 128
    if (idx >= 9 * 128 * 128) return;
    int lk = idx >> 14;
    int r  = idx & 16383;
    int n  = r >> 7;
    int k  = r & 127;
    wt[idx] = (__bf16)W[(lk << 14) + k * 128 + n];
}

// ------------------------- atom encoder ------------------------------------

__global__ void atom_enc_kernel(const int* __restrict__ x,
                                const float* __restrict__ atom_emb,  // [9][100][128]
                                float* __restrict__ h, int N) {
    long long idx = (long long)blockIdx.x * 256 + threadIdx.x;
    if (idx >= (long long)N * EMB) return;
    int n = (int)(idx >> 7);
    int j = (int)(idx & 127);
    const int* xr = x + n * 9;
    float s = 0.0f;
#pragma unroll
    for (int c = 0; c < 9; ++c)
        s += atom_emb[((c * 100) + xr[c]) * EMB + j];
    h[idx] = s;
}

// ------------------------- WMMA GEMM ---------------------------------------
// xw[kc] = h_bf16 @ W[layer,kc]  for kc = blockIdx.y in {0,1,2}.
// Block = 128 threads = 4 waves. Each block covers a 16-row strip x all 128
// cols; each wave owns 32 cols (two 16x16 C tiles), K-loop of 4 x 32.
__global__ void gemm_bf16_kernel(const __bf16* __restrict__ hbf,   // [M][128]
                                 const __bf16* __restrict__ wt,    // [3][128 n][128 k] (this layer)
                                 float* __restrict__ xw,           // [3][M*128]
                                 int M) {
    const int kc    = blockIdx.y;
    const int mtile = blockIdx.x;
    const int wave  = threadIdx.x >> 5;
    const int lane  = threadIdx.x & 31;
    const int half  = lane >> 4;     // 0: lanes 0-15, 1: lanes 16-31
    const int l15   = lane & 15;
    const int row0  = mtile * 16;
    const int n0    = wave * 32;

    // A: row (row0 + l15) of hbf, viewed as dword pairs of bf16.
    const unsigned* pa  = (const unsigned*)(hbf + (size_t)(row0 + l15) * EMB);
    // B: columns n0+l15 and n0+16+l15 of W (pre-transposed rows of wt).
    const unsigned* pb0 = (const unsigned*)(wt + ((size_t)kc * EMB + (n0 + l15)) * EMB);
    const unsigned* pb1 = (const unsigned*)(wt + ((size_t)kc * EMB + (n0 + 16 + l15)) * EMB);

    v8f acc0 = {};
    v8f acc1 = {};
#pragma unroll
    for (int kk = 0; kk < 4; ++kk) {          // K blocks of 32
        v8u au, bu0, bu1;
        // A-fragment: VGPR v holds K pair starting at
        //   half*16 + 2*v  within the 32-wide K block (matches B).
#pragma unroll
        for (int v = 0; v < 8; ++v)
            au[v] = pa[kk * 16 + half * 8 + v];
        // B-fragment: VGPR v holds K pair (half*16 + 2v) of column n -> 8
        // contiguous dwords of the transposed-W row.
        const int boff = kk * 16 + half * 8;
#pragma unroll
        for (int v = 0; v < 8; ++v) { bu0[v] = pb0[boff + v]; bu1[v] = pb1[boff + v]; }

        v16bf a  = __builtin_bit_cast(v16bf, au);
        v16bf b0 = __builtin_bit_cast(v16bf, bu0);
        v16bf b1 = __builtin_bit_cast(v16bf, bu1);
        acc0 = __builtin_amdgcn_wmma_f32_16x16x32_bf16(false, a, false, b0,
                                                       (short)0, acc0, false, false);
        acc1 = __builtin_amdgcn_wmma_f32_16x16x32_bf16(false, a, false, b1,
                                                       (short)0, acc1, false, false);
    }

    // C/D layout: VGPR v -> row (v + 8*half), lane&15 -> column.
    float* out = xw + (size_t)kc * M * EMB;
#pragma unroll
    for (int v = 0; v < 8; ++v) {
        int m = row0 + v + 8 * half;
        out[(size_t)m * EMB + n0 + l15]      = acc0[v];
        out[(size_t)m * EMB + n0 + 16 + l15] = acc1[v];
    }
}

// ------------------------- edge gather/scatter -----------------------------
// One thread per (edge, feature). Fuses the 3 bond-dim channels:
//   msg = sum_k ew[e,k] * xw_k[src[e], j]   -> single atomic add to agg[dst].
__global__ void edge_scatter_kernel(const int* __restrict__ src,
                                    const int* __restrict__ dst,
                                    const int* __restrict__ edge_attr,   // [E][3]
                                    const float* __restrict__ bond_emb_l, // [3][8][3]
                                    const float* __restrict__ xw,        // [3][M*128]
                                    float* __restrict__ agg, int E, int M) {
    long long gid = (long long)blockIdx.x * 256 + threadIdx.x;
    if (gid >= (long long)E * EMB) return;
    int e = (int)(gid >> 7);
    int j = (int)(gid & 127);

    float ew0 = 0.f, ew1 = 0.f, ew2 = 0.f;
#pragma unroll
    for (int c = 0; c < 3; ++c) {
        const float* p = bond_emb_l + (c * 8 + edge_attr[e * 3 + c]) * 3;
        ew0 += p[0]; ew1 += p[1]; ew2 += p[2];
    }

    size_t so = (size_t)src[e] * EMB + j;
    size_t NM = (size_t)M * EMB;
    float m = ew0 * xw[so] + ew1 * xw[NM + so] + ew2 * xw[2 * NM + so];
    unsafeAtomicAdd(&agg[(size_t)dst[e] * EMB + j], m);   // global_atomic_add_f32
}

// ------------------------- node update -------------------------------------

__global__ void node_update_kernel(float* __restrict__ h,
                                   const float* __restrict__ agg,
                                   const float* __restrict__ b_l,   // [3][128]
                                   int do_relu, int N) {
    long long idx = (long long)blockIdx.x * 256 + threadIdx.x;
    if (idx >= (long long)N * EMB) return;
    int j = (int)(idx & 127);
    float bs = b_l[j] + b_l[EMB + j] + b_l[2 * EMB + j];
    float v = agg[idx] + bs;
    if (do_relu) v = fmaxf(v, 0.0f);
    h[idx] = v + h[idx];
}

// ------------------------- mean pool + head --------------------------------

__global__ void pool_scatter_kernel(const float* __restrict__ h,
                                    const int* __restrict__ batch,
                                    float* __restrict__ sums,
                                    float* __restrict__ counts, int N) {
    long long idx = (long long)blockIdx.x * 256 + threadIdx.x;
    if (idx >= (long long)N * EMB) return;
    int n = (int)(idx >> 7);
    int j = (int)(idx & 127);
    int g = batch[n];
    unsafeAtomicAdd(&sums[(size_t)g * EMB + j], h[idx]);
    if (j == 0) unsafeAtomicAdd(&counts[g], 1.0f);
}

__global__ void head_kernel(const float* __restrict__ sums,
                            const float* __restrict__ counts,
                            const float* __restrict__ fc1_w,   // [128][128]
                            const float* __restrict__ fc1_b,   // [128]
                            const float* __restrict__ fc2_w,   // [128][1]
                            const float* __restrict__ fc2_b,   // [1]
                            float* __restrict__ out) {
    __shared__ float hg[EMB];
    __shared__ float red[EMB];
    int g = blockIdx.x;
    int j = threadIdx.x;

    float c = fmaxf(counts[g], 1.0f);
    hg[j] = sums[(size_t)g * EMB + j] / c;
    __syncthreads();

    float t = fc1_b[j];
#pragma unroll 8
    for (int i = 0; i < EMB; ++i)
        t += hg[i] * fc1_w[i * EMB + j];
    red[j] = t * fc2_w[j];
    __syncthreads();

    for (int s = 64; s > 0; s >>= 1) {
        if (j < s) red[j] += red[j + s];
        __syncthreads();
    }
    if (j == 0) out[g] = red[0] + fc2_b[0];
}

// ------------------------- host orchestration ------------------------------

extern "C" void kernel_launch(void* const* d_in, const int* in_sizes, int n_in,
                              void* d_out, int out_size, void* d_ws, size_t ws_size,
                              hipStream_t stream) {
    (void)in_sizes; (void)n_in; (void)out_size; (void)ws_size;

    const int*   x          = (const int*)d_in[0];     // [50000][9]
    const int*   edge_index = (const int*)d_in[1];     // [2][600000]
    const int*   edge_attr  = (const int*)d_in[2];     // [600000][3]
    const int*   batch      = (const int*)d_in[3];     // [50000]
    const float* atom_emb   = (const float*)d_in[4];   // [9][100][128]
    const float* bond_emb   = (const float*)d_in[5];   // [3][3][8][3]
    const float* W          = (const float*)d_in[6];   // [3][3][128][128]
    const float* b          = (const float*)d_in[7];   // [3][3][128]
    const float* fc1_w      = (const float*)d_in[8];   // [128][128]
    const float* fc1_b      = (const float*)d_in[9];   // [128]
    const float* fc2_w      = (const float*)d_in[10];  // [128][1]
    const float* fc2_b      = (const float*)d_in[11];  // [1]
    float* out = (float*)d_out;

    const int N = N_NODES, E = N_EDGES, G = N_GRAPHS;

    // Workspace carve-up (all chunks 256B-aligned by construction).
    char* ws = (char*)d_ws;
    float*  h      = (float*)ws;  ws += (size_t)N * EMB * 4;       // 25.6 MB
    __bf16* hbf    = (__bf16*)ws; ws += (size_t)N * EMB * 2;       // 12.8 MB
    __bf16* wt     = (__bf16*)ws; ws += (size_t)9 * EMB * EMB * 2; // 0.3 MB
    float*  xw     = (float*)ws;  ws += (size_t)3 * N * EMB * 4;   // 76.8 MB
    float*  agg    = (float*)ws;  ws += (size_t)N * EMB * 4;       // 25.6 MB
    float*  sums   = (float*)ws;  ws += (size_t)G * EMB * 4;       // 1.0 MB
    float*  counts = (float*)ws;  ws += (size_t)G * 4;             // 8 KB

    const long long NE = (long long)N * EMB;   // 6.4M
    const long long EE = (long long)E * EMB;   // 76.8M
    const unsigned gNE = (unsigned)((NE + 255) / 256);
    const unsigned gEE = (unsigned)((EE + 255) / 256);

    // Pre-transpose + convert all 9 weight matrices to bf16.
    prep_w_kernel<<<(9 * 128 * 128 + 255) / 256, 256, 0, stream>>>(W, wt);

    // Atom encoder: h = sum_c atom_emb[c][x[:,c]].
    atom_enc_kernel<<<gNE, 256, 0, stream>>>(x, atom_emb, h, N);

    for (int layer = 0; layer < 3; ++layer) {
        cvt_f32_to_bf16_kernel<<<gNE, 256, 0, stream>>>(h, hbf, NE);

        // xw[k] = h @ W[layer,k] via bf16 WMMA (M=50000 = 3125 * 16 exactly).
        gemm_bf16_kernel<<<dim3(N / 16, 3), 128, 0, stream>>>(
            hbf, wt + (size_t)layer * 3 * EMB * EMB, xw, N);

        zero_f32_kernel<<<gNE, 256, 0, stream>>>(agg, NE);

        edge_scatter_kernel<<<gEE, 256, 0, stream>>>(
            edge_index, edge_index + E, edge_attr,
            bond_emb + (size_t)layer * 3 * 8 * 3, xw, agg, E, N);

        node_update_kernel<<<gNE, 256, 0, stream>>>(
            h, agg, b + (size_t)layer * 3 * EMB, layer < 2 ? 1 : 0, N);
    }

    // Global mean pool.
    zero_f32_kernel<<<(G * EMB + 255) / 256, 256, 0, stream>>>(sums, (long long)G * EMB);
    zero_f32_kernel<<<(G + 255) / 256, 256, 0, stream>>>(counts, (long long)G);
    pool_scatter_kernel<<<gNE, 256, 0, stream>>>(h, batch, sums, counts, N);

    // Head MLP.
    head_kernel<<<G, EMB, 0, stream>>>(sums, counts, fc1_w, fc1_b, fc2_w, fc2_b, out);
}
